// RTEncoderOnly_8160437862432
// MI455X (gfx1250) — matrix operand order repaired and checked by reading, not compiled, not run on hardware
//
#include <hip/hip_runtime.h>

// ---------------------------------------------------------------------------
// Model constants (reference: B,S,D,H,L,FF,TEXT = 4,1024,1024,16,4,4096,384)
// ---------------------------------------------------------------------------
constexpr int BATCH  = 4;
constexpr int SEQ    = 1024;
constexpr int DIM    = 1024;
constexpr int HEADS  = 16;
constexpr int HDIM   = 64;
constexpr int FFD    = 4096;
constexpr int TXT    = 384;
constexpr int LAYERS = 4;
constexpr int TOK    = BATCH * SEQ;   // 4096

#define GF_BIAS 1
#define GF_GELU 2
#define GF_RES  4

typedef __attribute__((ext_vector_type(8)))  float  v8f;
typedef __attribute__((ext_vector_type(16))) __bf16 v16bf;
typedef __attribute__((ext_vector_type(8)))  __bf16 v8bf;

__device__ __forceinline__ v16bf cat8(v8bf lo, v8bf hi) {
  return __builtin_shufflevector(lo, hi, 0,1,2,3,4,5,6,7,8,9,10,11,12,13,14,15);
}
__device__ __forceinline__ v8bf cvt8(float4 a, float4 b) {
  v8bf r;
  r[0]=(__bf16)a.x; r[1]=(__bf16)a.y; r[2]=(__bf16)a.z; r[3]=(__bf16)a.w;
  r[4]=(__bf16)b.x; r[5]=(__bf16)b.y; r[6]=(__bf16)b.z; r[7]=(__bf16)b.w;
  return r;
}

// gfx1250 async memory->LDS copy (16B per lane). Per CDNA5 ISA the
// instruction offset is added to BOTH the LDS and the global address, so one
// LDS-address VGPR + one 64-bit global-address VGPR pair serves both chunks.
__device__ __forceinline__ void async_lds_b128(unsigned lds_addr, const void* gaddr) {
  asm volatile("global_load_async_to_lds_b128 %0, %1, off"
               :: "v"(lds_addr), "v"(gaddr) : "memory");
}
__device__ __forceinline__ void async_lds_b128_off16(unsigned lds_addr, const void* gaddr) {
  asm volatile("global_load_async_to_lds_b128 %0, %1, off offset:16"
               :: "v"(lds_addr), "v"(gaddr) : "memory");
}
__device__ __forceinline__ void wait_asynccnt0() {
  asm volatile("s_wait_asynccnt 0x0" ::: "memory");
}

// ---------------------------------------------------------------------------
// Weight convert: W (K x N, fp32, row major) -> Wt (N x K, bf16)  so B-matrix
// WMMA fragments (lane = N column, contiguous K) are two aligned 16B loads.
// ---------------------------------------------------------------------------
__global__ void convert_wt_kernel(const float* __restrict__ W, __bf16* __restrict__ Wt,
                                  int K, int N) {
  int idx = blockIdx.x * 256 + threadIdx.x;
  if (idx >= K * N) return;
  int n = idx / K, k = idx - n * K;
  Wt[idx] = (__bf16)W[(size_t)k * N + n];
}

// ---------------------------------------------------------------------------
// Mask precompute: 4 mask types packed 32 keys per u32 word.
// bits[((b*4+type)*SEQ + q)*32 + kword]
// ---------------------------------------------------------------------------
__global__ void build_masks_kernel(const int* __restrict__ node, const int* __restrict__ f2p,
                                   const int* __restrict__ colx, const int* __restrict__ tabx,
                                   const unsigned char* __restrict__ pad,
                                   unsigned* __restrict__ bits) {
  const int b = blockIdx.y, q = blockIdx.x, t = threadIdx.x;   // 32 threads
  const int tokq = b * SEQ + q;
  const bool vq = !pad[tokq];
  const int nq = node[tokq];
  int fq[5];
#pragma unroll
  for (int i = 0; i < 5; i++) fq[i] = f2p[tokq * 5 + i];
  const int cq = colx[tokq], tq = tabx[tokq];
  unsigned wf = 0, wn = 0, wc = 0, wl = 0;
  for (int i = 0; i < 32; i++) {
    const int k = t * 32 + i;
    const int tokk = b * SEQ + k;
    const bool p = vq && !pad[tokk];
    const int nk = node[tokk];
    bool same = (nk == nq);
    bool kvin = false, qin = false;
#pragma unroll
    for (int j = 0; j < 5; j++) {
      kvin |= (nk == fq[j]);
      qin  |= (nq == f2p[tokk * 5 + j]);
    }
    bool ct = (colx[tokk] == cq) && (tabx[tokk] == tq);
    unsigned m = 1u << i;
    if (p && (same || kvin)) wf |= m;
    if (p && qin)            wn |= m;
    if (p && ct)             wc |= m;
    if (p)                   wl |= m;
  }
  size_t base = (((size_t)b * 4) * SEQ + q) * 32 + t;
  bits[base + 0 * (size_t)SEQ * 32] = wf;
  bits[base + 1 * (size_t)SEQ * 32] = wn;
  bits[base + 2 * (size_t)SEQ * 32] = wc;
  bits[base + 3 * (size_t)SEQ * 32] = wl;
}

// ---------------------------------------------------------------------------
// Generic WMMA GEMM: C(MxN) = act(A(MxK,fp32->bf16) * Wt(NxK,bf16) + bias) + res
// 128x128 tile / workgroup, 8 waves, each wave 32x64 (2x4 WMMA tiles).
// A tile: VGPR path (needs fp32->bf16 convert).  B tile (already bf16 in
// global): async memory->LDS path tracked by ASYNCcnt.
// ---------------------------------------------------------------------------
__global__ __launch_bounds__(256)
void gemm_bf16_kernel(const float* __restrict__ A, const __bf16* __restrict__ Wt,
                      const float* __restrict__ bias, const float* __restrict__ res,
                      float* __restrict__ C, int M, int N, int K, int flags) {
  __shared__ __attribute__((aligned(16))) __bf16 lA[128][40];  // 32 K cols + pad (80B stride)
  __shared__ __attribute__((aligned(16))) __bf16 lB[128][40];
  const int t = threadIdx.x;
  const int lane = t & 31, wv = t >> 5;
  const int half = lane >> 4, l15 = lane & 15;
  const int m0 = blockIdx.y * 128, n0 = blockIdx.x * 128;
  const int mw = (wv >> 1) * 32, nw = (wv & 1) * 64;

  v8f acc[2][4];
#pragma unroll
  for (int mi = 0; mi < 2; mi++)
#pragma unroll
    for (int ni = 0; ni < 4; ni++)
#pragma unroll
      for (int r = 0; r < 8; r++) acc[mi][ni][r] = 0.0f;

  const int rA = t >> 1, cA = (t & 1) * 16;
  const unsigned ldsB = (unsigned)(uintptr_t)&lB[rA][cA];
  for (int k0 = 0; k0 < K; k0 += 32) {
    {
      // B tile: async global->LDS (bf16 weights, no conversion needed)
      const __bf16* gB = Wt + (size_t)(n0 + rA) * K + k0 + cA;
      async_lds_b128(ldsB, gB);          // 16B: K = cA..cA+7
      async_lds_b128_off16(ldsB, gB);    // 16B: K = cA+8..cA+15 (offset on both sides)
      // A tile: load fp32, convert to bf16, store to LDS
      const float* gA = A + (size_t)(m0 + rA) * K + k0 + cA;
      float4 f0 = ((const float4*)gA)[0], f1 = ((const float4*)gA)[1];
      float4 f2 = ((const float4*)gA)[2], f3 = ((const float4*)gA)[3];
      *(v8bf*)&lA[rA][cA]     = cvt8(f0, f1);
      *(v8bf*)&lA[rA][cA + 8] = cvt8(f2, f3);
      if (k0 + 32 < K) {                 // gfx1250 global_prefetch_b8 path
        __builtin_prefetch(gA + 32, 0, 1);
        __builtin_prefetch(gB + 32, 0, 1);
      }
    }
    wait_asynccnt0();                    // own wave's async copies landed in LDS
    __syncthreads();                     // all waves' tiles visible
    v16bf af[2], bfr[4];
#pragma unroll
    for (int mi = 0; mi < 2; mi++) {
      const int row = mw + mi * 16 + l15;      // A frag: lane = M row, K {kh*8, 16+kh*8}
      af[mi] = cat8(*(const v8bf*)&lA[row][half * 8],
                    *(const v8bf*)&lA[row][16 + half * 8]);
    }
#pragma unroll
    for (int ni = 0; ni < 4; ni++) {
      const int row = nw + ni * 16 + l15;      // B frag: lane = N col, K {half*8, 16+half*8}
      bfr[ni] = cat8(*(const v8bf*)&lB[row][half * 8],
                     *(const v8bf*)&lB[row][16 + half * 8]);
    }
#pragma unroll
    for (int mi = 0; mi < 2; mi++)
#pragma unroll
      for (int ni = 0; ni < 4; ni++)
        acc[mi][ni] = __builtin_amdgcn_wmma_f32_16x16x32_bf16(
            false, af[mi], false, bfr[ni], (short)0, acc[mi][ni], false, false);
    __syncthreads();
  }

#pragma unroll
  for (int mi = 0; mi < 2; mi++)
#pragma unroll
    for (int ni = 0; ni < 4; ni++)
#pragma unroll
      for (int r = 0; r < 8; r++) {
        const int row = m0 + mw + mi * 16 + r + half * 8;   // C layout: m=r+8*half
        const int col = n0 + nw + ni * 16 + l15;            //           n=lane&15
        float v = acc[mi][ni][r];
        if (flags & GF_BIAS) v += bias[col];
        if (flags & GF_GELU) {
          float u = v;
          v = 0.5f * u * (1.0f + tanhf(0.79788456f * (u + 0.044715f * u * u * u)));
        }
        if (flags & GF_RES) v += res[(size_t)row * N + col];
        C[(size_t)row * N + col] = v;
      }
}

// ---------------------------------------------------------------------------
// Flash attention: workgroup = (64 q rows, head h, batch b); 4 waves x 16 rows.
// Online softmax with 16-lane butterfly reductions; both GEMMs are WMMA bf16.
// ---------------------------------------------------------------------------
__global__ __launch_bounds__(128)
void attention_kernel(const float* __restrict__ Q, const float* __restrict__ Kg,
                      const float* __restrict__ V, const unsigned* __restrict__ bits,
                      float* __restrict__ O) {
  __shared__ __attribute__((aligned(16))) __bf16 lQ[64][72];     // 64 hd + pad (144B)
  __shared__ __attribute__((aligned(16))) __bf16 lK[32][72];
  __shared__ __attribute__((aligned(16))) __bf16 lVt[64][40];    // V^T: [hd][key]
  __shared__ __attribute__((aligned(16))) __bf16 lP[4][16][40];  // per-wave P tile
  const int t = threadIdx.x;
  const int lane = t & 31, wv = t >> 5;
  const int half = lane >> 4, l15 = lane & 15;
  const int qb = blockIdx.x * 64, h = blockIdx.y, b = blockIdx.z;
  const int mt = h >> 2;                       // mask type: 4 heads per type

  {  // stage Q block 64x64 fp32 -> bf16
    const int r = t >> 1, c0 = (t & 1) * 32;
    const float* g = Q + ((size_t)(b * SEQ + qb + r) * DIM) + h * HDIM + c0;
#pragma unroll
    for (int u = 0; u < 2; u++) {
      float4 f0 = ((const float4*)g)[u * 4 + 0], f1 = ((const float4*)g)[u * 4 + 1];
      float4 f2 = ((const float4*)g)[u * 4 + 2], f3 = ((const float4*)g)[u * 4 + 3];
      *(v8bf*)&lQ[r][c0 + u * 16]     = cvt8(f0, f1);
      *(v8bf*)&lQ[r][c0 + u * 16 + 8] = cvt8(f2, f3);
    }
  }
  __syncthreads();
  v16bf aQ[2];
#pragma unroll
  for (int c = 0; c < 2; c++) {
    const int row = wv * 16 + l15;
    aQ[c] = cat8(*(const v8bf*)&lQ[row][c * 32 + half * 8],
                 *(const v8bf*)&lQ[row][c * 32 + 16 + half * 8]);
  }

  v8f Oa[4];
#pragma unroll
  for (int jo = 0; jo < 4; jo++)
#pragma unroll
    for (int r = 0; r < 8; r++) Oa[jo][r] = 0.0f;
  float mrun[8], lrun[8];
  int any[8];
#pragma unroll
  for (int r = 0; r < 8; r++) { mrun[r] = -3.0e38f; lrun[r] = 0.0f; any[r] = 0; }

  for (int kbk = 0; kbk < SEQ / 32; kbk++) {
    __syncthreads();
    {  // stage K (32x64) and V^T (64x32)
      const int r = t >> 2, c0 = (t & 3) * 16;
      const float* gk = Kg + ((size_t)(b * SEQ + kbk * 32 + r) * DIM) + h * HDIM + c0;
      float4 f0 = ((const float4*)gk)[0], f1 = ((const float4*)gk)[1];
      float4 f2 = ((const float4*)gk)[2], f3 = ((const float4*)gk)[3];
      *(v8bf*)&lK[r][c0]     = cvt8(f0, f1);
      *(v8bf*)&lK[r][c0 + 8] = cvt8(f2, f3);
      const float* gv = V + ((size_t)(b * SEQ + kbk * 32 + r) * DIM) + h * HDIM + c0;
      f0 = ((const float4*)gv)[0]; f1 = ((const float4*)gv)[1];
      f2 = ((const float4*)gv)[2]; f3 = ((const float4*)gv)[3];
      float fv[16] = {f0.x,f0.y,f0.z,f0.w, f1.x,f1.y,f1.z,f1.w,
                      f2.x,f2.y,f2.z,f2.w, f3.x,f3.y,f3.z,f3.w};
#pragma unroll
      for (int i = 0; i < 16; i++) lVt[c0 + i][r] = (__bf16)fv[i];
    }
    __syncthreads();

    float s[8][2];
#pragma unroll
    for (int j = 0; j < 2; j++) {  // scores: Q(16x64) * K^T -> 16x16 per j
      v8f accs;
#pragma unroll
      for (int r = 0; r < 8; r++) accs[r] = 0.0f;
#pragma unroll
      for (int c = 0; c < 2; c++) {
        const int row = j * 16 + l15;
        v16bf bK = cat8(*(const v8bf*)&lK[row][c * 32 + half * 8],
                        *(const v8bf*)&lK[row][c * 32 + 16 + half * 8]);
        accs = __builtin_amdgcn_wmma_f32_16x16x32_bf16(false, aQ[c], false, bK,
                                                       (short)0, accs, false, false);
      }
#pragma unroll
      for (int r = 0; r < 8; r++) s[r][j] = accs[r] * 0.125f;  // 1/sqrt(64)
    }

    float alpha[8];
#pragma unroll
    for (int r = 0; r < 8; r++) {
      const int qrow = qb + wv * 16 + r + half * 8;
      const unsigned w = bits[(((size_t)b * 4 + mt) * SEQ + qrow) * 32 + kbk];
#pragma unroll
      for (int j = 0; j < 2; j++)
        if (!((w >> (j * 16 + l15)) & 1u)) s[r][j] = -1.0e9f;
      float tmax = fmaxf(s[r][0], s[r][1]);
      for (int d = 1; d < 16; d <<= 1) tmax = fmaxf(tmax, __shfl_xor(tmax, d, 32));
      const float mnew = fmaxf(mrun[r], tmax);
      alpha[r] = __expf(mrun[r] - mnew);
      const float p0 = __expf(s[r][0] - mnew), p1 = __expf(s[r][1] - mnew);
      float rsum = p0 + p1;
      for (int d = 1; d < 16; d <<= 1) rsum += __shfl_xor(rsum, d, 32);
      lrun[r] = lrun[r] * alpha[r] + rsum;
      mrun[r] = mnew;
      if (tmax > -1.0e8f) any[r] = 1;
      lP[wv][r + half * 8][l15]      = (__bf16)p0;   // P into A-fragment layout
      lP[wv][r + half * 8][16 + l15] = (__bf16)p1;
    }
#pragma unroll
    for (int jo = 0; jo < 4; jo++)
#pragma unroll
      for (int r = 0; r < 8; r++) Oa[jo][r] *= alpha[r];
    __syncthreads();

    v16bf aP = cat8(*(const v8bf*)&lP[wv][l15][half * 8],
                    *(const v8bf*)&lP[wv][l15][16 + half * 8]);
#pragma unroll
    for (int jo = 0; jo < 4; jo++) {  // O += P(16x32) * V(32x64)
      const int row = jo * 16 + l15;
      v16bf bV = cat8(*(const v8bf*)&lVt[row][half * 8],
                      *(const v8bf*)&lVt[row][16 + half * 8]);
      Oa[jo] = __builtin_amdgcn_wmma_f32_16x16x32_bf16(false, aP, false, bV,
                                                       (short)0, Oa[jo], false, false);
    }
  }

#pragma unroll
  for (int r = 0; r < 8; r++) {
    const float inv = any[r] ? (1.0f / lrun[r]) : 0.0f;
    const int qrow = qb + wv * 16 + r + half * 8;
    float* op = O + ((size_t)(b * SEQ + qrow) * DIM) + h * HDIM;
#pragma unroll
    for (int jo = 0; jo < 4; jo++) op[jo * 16 + l15] = Oa[jo][r] * inv;
  }
}

// ---------------------------------------------------------------------------
// Block LayerNorm helpers
// ---------------------------------------------------------------------------
__device__ __forceinline__ void block_stats(float* rs, float* rs2, int t, float s, float s2,
                                            float& mean, float& rstd) {
  __syncthreads();
  rs[t] = s; rs2[t] = s2;
  __syncthreads();
  for (int d = 128; d > 0; d >>= 1) {
    if (t < d) { rs[t] += rs[t + d]; rs2[t] += rs2[t + d]; }
    __syncthreads();
  }
  mean = rs[0] * (1.0f / 1024.0f);
  float var = rs2[0] * (1.0f / 1024.0f) - mean * mean;
  rstd = rsqrtf(var + 1e-5f);
}

__global__ __launch_bounds__(256)
void ln_kernel(const float* __restrict__ in, float* __restrict__ out,
               const float* __restrict__ g, const float* __restrict__ bta) {
  __shared__ float rs[256], rs2[256];
  const int tok = blockIdx.x, t = threadIdx.x;
  const float* p = in + (size_t)tok * DIM;
  float v[4]; float s = 0, s2 = 0;
#pragma unroll
  for (int i = 0; i < 4; i++) { v[i] = p[t + 256 * i]; s += v[i]; s2 += v[i] * v[i]; }
  float mean, rstd;
  block_stats(rs, rs2, t, s, s2, mean, rstd);
#pragma unroll
  for (int i = 0; i < 4; i++) {
    int j = t + 256 * i;
    out[(size_t)tok * DIM + j] = (v[i] - mean) * rstd * g[j] + bta[j];
  }
}

// ---------------------------------------------------------------------------
// Embedding combine: x = LN(Ecol)*valid + LN(stream[sem])*obs + mask_emb*msk
// ---------------------------------------------------------------------------
__global__ __launch_bounds__(256)
void build_x_kernel(const float* __restrict__ Ecol, const float* __restrict__ Etext,
                    const float* gC, const float* bC, const float* gT, const float* bT,
                    const float* gN, const float* bN, const float* gD, const float* bD,
                    const float* gB, const float* bB,
                    const float* wN, const float* biN, const float* wD, const float* biD,
                    const float* wB, const float* biB,
                    const float* meN, const float* meT, const float* meD, const float* meB,
                    const int* __restrict__ sem, const unsigned char* __restrict__ pad,
                    const unsigned char* __restrict__ mk,
                    const float* __restrict__ numv, const float* __restrict__ datev,
                    const float* __restrict__ boolv, float* __restrict__ x) {
  __shared__ float rs[256], rs2[256];
  const int tok = blockIdx.x, t = threadIdx.x;
  const bool valid = !pad[tok];
  const int st = sem[tok];
  const bool mskd = valid && mk[tok];
  const bool obs = valid && !mk[tok];
  float e[4]; float s = 0, s2 = 0;
#pragma unroll
  for (int i = 0; i < 4; i++) {
    e[i] = Ecol[(size_t)tok * DIM + t + 256 * i];
    s += e[i]; s2 += e[i] * e[i];
  }
  float mean, rstd;
  block_stats(rs, rs2, t, s, s2, mean, rstd);
  float xv[4];
#pragma unroll
  for (int i = 0; i < 4; i++) {
    int j = t + 256 * i;
    xv[i] = valid ? ((e[i] - mean) * rstd * gC[j] + bC[j]) : 0.0f;
  }
  const float *ng, *nb, *me, *ew = nullptr, *eb = nullptr;
  float sval = 0.0f;
  if (st == 0)      { ng = gN; nb = bN; me = meN; ew = wN; eb = biN; sval = numv[tok]; }
  else if (st == 1) { ng = gT; nb = bT; me = meT; }
  else if (st == 2) { ng = gD; nb = bD; me = meD; ew = wD; eb = biD; sval = datev[tok]; }
  else              { ng = gB; nb = bB; me = meB; ew = wB; eb = biB; sval = boolv[tok]; }
  float f[4]; s = 0; s2 = 0;
#pragma unroll
  for (int i = 0; i < 4; i++) {
    int j = t + 256 * i;
    float v = (st == 1) ? Etext[(size_t)tok * DIM + j] : (sval * ew[j] + eb[j]);
    f[i] = v; s += v; s2 += v * v;
  }
  block_stats(rs, rs2, t, s, s2, mean, rstd);
#pragma unroll
  for (int i = 0; i < 4; i++) {
    int j = t + 256 * i;
    float v = xv[i];
    if (obs)  v += (f[i] - mean) * rstd * ng[j] + nb[j];
    if (mskd) v += me[j];
    x[(size_t)tok * DIM + j] = v;
  }
}

// ---------------------------------------------------------------------------
// Host launch. Param leaf order assumes jax tree flatten (dict keys sorted):
//  blocks[l]: W1,W2,Wk,Wo,Wq,Wv,b1,b2,ln1_b,ln1_g,ln2_b,ln2_g  (12..59)
//  enc: boolean W,b | col_name W,b | datetime W,b | number W,b | text W,b (60..69)
//  mask_embs: boolean, datetime, number, text (70..73)
//  norm: (b,g) x boolean,col_name,datetime,number,text (74..83)
//  out_b=84, out_g=85
// ---------------------------------------------------------------------------
extern "C" void kernel_launch(void* const* d_in, const int* in_sizes, int n_in,
                              void* d_out, int out_size, void* d_ws, size_t ws_size,
                              hipStream_t stream) {
  (void)in_sizes; (void)n_in; (void)out_size; (void)ws_size;
  const int* node = (const int*)d_in[0];
  const int* f2p  = (const int*)d_in[1];
  const int* colx = (const int*)d_in[2];
  const int* tabx = (const int*)d_in[3];
  const int* sem  = (const int*)d_in[4];
  const unsigned char* pad = (const unsigned char*)d_in[5];
  const unsigned char* mk  = (const unsigned char*)d_in[6];
  const float* numv  = (const float*)d_in[7];
  const float* txtv  = (const float*)d_in[8];
  const float* datev = (const float*)d_in[9];
  const float* boolv = (const float*)d_in[10];
  const float* colv  = (const float*)d_in[11];
  auto blkP = [&](int l, int j) { return (const float*)d_in[12 + l * 12 + j]; };
  const float* encBoolW = (const float*)d_in[60];
  const float* encBoolB = (const float*)d_in[61];
  const float* encColW  = (const float*)d_in[62];
  const float* encColB  = (const float*)d_in[63];
  const float* encDateW = (const float*)d_in[64];
  const float* encDateB = (const float*)d_in[65];
  const float* encNumW  = (const float*)d_in[66];
  const float* encNumB  = (const float*)d_in[67];
  const float* encTxtW  = (const float*)d_in[68];
  const float* encTxtB  = (const float*)d_in[69];
  const float* meBool = (const float*)d_in[70];
  const float* meDate = (const float*)d_in[71];
  const float* meNum  = (const float*)d_in[72];
  const float* meTxt  = (const float*)d_in[73];
  const float* nBoolB = (const float*)d_in[74];
  const float* nBoolG = (const float*)d_in[75];
  const float* nColB  = (const float*)d_in[76];
  const float* nColG  = (const float*)d_in[77];
  const float* nDateB = (const float*)d_in[78];
  const float* nDateG = (const float*)d_in[79];
  const float* nNumB  = (const float*)d_in[80];
  const float* nNumG  = (const float*)d_in[81];
  const float* nTxtB  = (const float*)d_in[82];
  const float* nTxtG  = (const float*)d_in[83];
  const float* outB = (const float*)d_in[84];
  const float* outG = (const float*)d_in[85];

  char* ws = (char*)d_ws;
  size_t off = 0;
  auto take = [&](size_t bytes) -> char* {
    char* p = ws + off;
    off += (bytes + 255) & ~(size_t)255;
    return p;
  };
  __bf16* wcolT = (__bf16*)take((size_t)DIM * TXT * 2);
  __bf16* wtxtT = (__bf16*)take((size_t)DIM * TXT * 2);
  __bf16 *wqT[LAYERS], *wkT[LAYERS], *wvT[LAYERS], *woT[LAYERS], *w1T[LAYERS], *w2T[LAYERS];
  for (int l = 0; l < LAYERS; l++) {
    wqT[l] = (__bf16*)take((size_t)DIM * DIM * 2);
    wkT[l] = (__bf16*)take((size_t)DIM * DIM * 2);
    wvT[l] = (__bf16*)take((size_t)DIM * DIM * 2);
    woT[l] = (__bf16*)take((size_t)DIM * DIM * 2);
    w1T[l] = (__bf16*)take((size_t)FFD * DIM * 2);
    w2T[l] = (__bf16*)take((size_t)DIM * FFD * 2);
  }
  unsigned* maskbits = (unsigned*)take((size_t)BATCH * 4 * SEQ * 32 * 4);
  float* Ecol = (float*)take((size_t)TOK * DIM * 4);
  float* Etxt = (float*)take((size_t)TOK * DIM * 4);
  float* xb   = (float*)take((size_t)TOK * DIM * 4);
  float* hb   = (float*)take((size_t)TOK * DIM * 4);
  float* qb   = (float*)take((size_t)TOK * DIM * 4);
  float* kb   = (float*)take((size_t)TOK * DIM * 4);
  float* vb   = (float*)take((size_t)TOK * DIM * 4);
  float* ob   = (float*)take((size_t)TOK * DIM * 4);
  float* ffn  = (float*)take((size_t)TOK * FFD * 4);

  auto conv = [&](const float* W, __bf16* Wt, int K, int N) {
    int n = K * N;
    convert_wt_kernel<<<dim3((n + 255) / 256), dim3(256), 0, stream>>>(W, Wt, K, N);
  };
  conv(encColW, wcolT, TXT, DIM);
  conv(encTxtW, wtxtT, TXT, DIM);
  for (int l = 0; l < LAYERS; l++) {
    conv(blkP(l, 4), wqT[l], DIM, DIM);   // Wq
    conv(blkP(l, 2), wkT[l], DIM, DIM);   // Wk
    conv(blkP(l, 5), wvT[l], DIM, DIM);   // Wv
    conv(blkP(l, 3), woT[l], DIM, DIM);   // Wo
    conv(blkP(l, 0), w1T[l], DIM, FFD);   // W1
    conv(blkP(l, 1), w2T[l], FFD, DIM);   // W2
  }
  build_masks_kernel<<<dim3(SEQ, BATCH), dim3(32), 0, stream>>>(node, f2p, colx, tabx, pad,
                                                                maskbits);
  auto gemm = [&](const float* A, const __bf16* Wt, const float* bias, const float* res,
                  float* C, int M_, int N_, int K_, int flags) {
    gemm_bf16_kernel<<<dim3(N_ / 128, M_ / 128), dim3(256), 0, stream>>>(A, Wt, bias, res, C,
                                                                         M_, N_, K_, flags);
  };
  gemm(colv, wcolT, encColB, nullptr, Ecol, TOK, DIM, TXT, GF_BIAS);
  gemm(txtv, wtxtT, encTxtB, nullptr, Etxt, TOK, DIM, TXT, GF_BIAS);
  build_x_kernel<<<dim3(TOK), dim3(256), 0, stream>>>(
      Ecol, Etxt, nColG, nColB, nTxtG, nTxtB, nNumG, nNumB, nDateG, nDateB, nBoolG, nBoolB,
      encNumW, encNumB, encDateW, encDateB, encBoolW, encBoolB,
      meNum, meTxt, meDate, meBool, sem, pad, mk, numv, datev, boolv, xb);

  for (int l = 0; l < LAYERS; l++) {
    ln_kernel<<<dim3(TOK), dim3(256), 0, stream>>>(xb, hb, blkP(l, 9), blkP(l, 8));
    gemm(hb, wqT[l], nullptr, nullptr, qb, TOK, DIM, DIM, 0);
    gemm(hb, wkT[l], nullptr, nullptr, kb, TOK, DIM, DIM, 0);
    gemm(hb, wvT[l], nullptr, nullptr, vb, TOK, DIM, DIM, 0);
    attention_kernel<<<dim3(SEQ / 64, HEADS, BATCH), dim3(128), 0, stream>>>(qb, kb, vb,
                                                                             maskbits, ob);
    gemm(ob, woT[l], nullptr, xb, xb, TOK, DIM, DIM, GF_RES);
    ln_kernel<<<dim3(TOK), dim3(256), 0, stream>>>(xb, hb, blkP(l, 11), blkP(l, 10));
    gemm(hb, w1T[l], blkP(l, 6), nullptr, ffn, TOK, FFD, DIM, GF_BIAS | GF_GELU);
    gemm(ffn, w2T[l], blkP(l, 7), xb, xb, TOK, DIM, FFD, GF_BIAS | GF_RES);
  }
  ln_kernel<<<dim3(TOK), dim3(256), 0, stream>>>(xb, (float*)d_out, outG, outB);
}
